// GRUModel_89223650607207
// MI455X (gfx1250) — matrix operand order repaired and checked by reading, not compiled, not run on hardware
//
#include <hip/hip_runtime.h>
#include <hip/hip_bf16.h>

// ---------------- CDNA5 (gfx1250) WMMA helpers -------------------------------
typedef unsigned short u16;
typedef __bf16 bf16;
typedef bf16  v16bf __attribute__((ext_vector_type(16)));
typedef float v8f   __attribute__((ext_vector_type(8)));

#define B_   64
#define L_   720
#define C_   321
#define SEG_ 48
#define D_   256
#define SX_  15
#define SY_  15
#define BC_  (B_ * C_)        // 20544 = 1284 * 16
#define G3_  (3 * D_)         // 768

__device__ __forceinline__ u16 f2bf(float f) {
    unsigned u = __float_as_uint(f);
    u += 0x7fffu + ((u >> 16) & 1u);          // round-to-nearest-even
    return (u16)(u >> 16);
}
__device__ __forceinline__ float sigmoidf_(float x) { return 1.0f / (1.0f + __expf(-x)); }

union Frag16 { uint4 u[2]; v16bf v; };

// A: 16x32 bf16 tile, row-major source with leading dim `ld` (elements).
// Lane l: M = l&15; lanes<16 hold K {kb..kb+7, kb+16..kb+23}, lanes>=16 hold
// K {kb+8..kb+15, kb+24..kb+31}  (ISA 7.12.2, 16-bit A layout).
__device__ __forceinline__ v16bf load_a16(const u16* base, int ld, int kb, int lane) {
    const int m = lane & 15, kh = (lane >> 4) & 1;
    const u16* p = base + (size_t)m * ld + kb + kh * 8;
    Frag16 f;
    f.u[0] = *(const uint4*)(p);
    f.u[1] = *(const uint4*)(p + 16);
    return f.v;
}
// B: 32x16 bf16 tile where B(k,n) = W[(col0+n)*ld + k], W row-major (out x K).
// Lane l: N = l&15; lanes<16 hold K {kb..kb+7, kb+16..kb+23}, lanes>=16 hold
// K {kb+8..kb+15, kb+24..kb+31} (same K order as A -> two global_load_b128 per lane).
__device__ __forceinline__ v16bf load_b16(const u16* W, int ld, int col0, int kb, int lane) {
    const int n = lane & 15, kh = (lane >> 4) & 1;
    const u16* p = W + (size_t)(col0 + n) * ld + kb + kh * 8;
    Frag16 f;
    f.u[0] = *(const uint4*)(p);
    f.u[1] = *(const uint4*)(p + 16);
    return f.v;
}
#define WMMA_BF16(a, b, c) \
    __builtin_amdgcn_wmma_f32_16x16x32_bf16(false, (a), false, (b), (short)0, (c), false, false)

// Async DMA: move one 16-byte chunk global -> LDS (tracked by ASYNCcnt).
// VDST vgpr carries the workgroup-relative LDS byte offset (== low 32 bits of
// the generic pointer to a __shared__ object), VADDR the 64-bit global addr.
__device__ __forceinline__ void async_g2l_b128(const void* gaddr, void* lds) {
    const unsigned long long ga = (unsigned long long)gaddr;
    const unsigned lo = (unsigned)(size_t)lds;
    asm volatile("global_load_async_to_lds_b128 %0, %1, off"
                 :: "v"(lo), "v"(ga) : "memory");
}
__device__ __forceinline__ void wait_async0() {
#if __has_builtin(__builtin_amdgcn_s_wait_asynccnt)
    __builtin_amdgcn_s_wait_asynccnt(0);
#else
    asm volatile("s_wait_asynccnt 0x0" ::: "memory");
#endif
}

// ---------------- kernel 0: weight f32->bf16 + decoder tokens ----------------
__global__ void prep_kernel(const float* __restrict__ Wih,  const float* __restrict__ Whh,
                            const float* __restrict__ Wres, const float* __restrict__ gWih,
                            const float* __restrict__ gWhh, const float* __restrict__ Wpred,
                            const float* __restrict__ pos_emb, const float* __restrict__ chan_emb,
                            u16* Wih_bf, u16* Whh_bf, u16* Wres_bf,
                            u16* gWih_bf, u16* gWhh_bf, u16* Wpred_bf, u16* tok_bf) {
    const int N0 = G3_ * D_, N1 = D_ * D_, N2 = SEG_ * D_, NT = 4816 * D_;
    const int total = 4 * N0 + N1 + N2 + NT;
    for (int i = blockIdx.x * blockDim.x + threadIdx.x; i < total; i += gridDim.x * blockDim.x) {
        int j = i;
        if (j < N0) { Wih_bf [j] = f2bf(Wih [j]); continue; } j -= N0;
        if (j < N0) { Whh_bf [j] = f2bf(Whh [j]); continue; } j -= N0;
        if (j < N1) { Wres_bf[j] = f2bf(Wres[j]); continue; } j -= N1;
        if (j < N0) { gWih_bf[j] = f2bf(gWih[j]); continue; } j -= N0;
        if (j < N0) { gWhh_bf[j] = f2bf(gWhh[j]); continue; } j -= N0;
        if (j < N2) { Wpred_bf[j] = f2bf(Wpred[j]); continue; } j -= N2;
        // decoder tokens: row = c*SY + sy, [pos_emb[sy] | channel_emb[c]]; pad row 4815 = 0
        const int row = j >> 8, d = j & 255;
        float v = 0.0f;
        if (row < C_ * SY_) {
            const int c = row / SY_, sy = row % SY_;
            v = (d < 128) ? pos_emb[sy * 128 + d] : chan_emb[c * 128 + (d - 128)];
        }
        tok_bf[j] = f2bf(v);
    }
}

__global__ void zero_kernel(uint4* p, int n) {
    const uint4 z = {0u, 0u, 0u, 0u};
    for (int i = blockIdx.x * blockDim.x + threadIdx.x; i < n; i += gridDim.x * blockDim.x)
        p[i] = z;
}

// ---------------- kernel 1: fused encoder GRU step (launched SX_=15 times) ---
// Block = 16 rows (one (b,c) strip), 256 threads = 8 waves.
//   carry: h_new = xt + Wres * GRUCell(h, xt) + bres
__global__ void __launch_bounds__(256) step_kernel(
        const float* __restrict__ x, const float* __restrict__ Wemb, const float* __restrict__ bemb,
        const u16* __restrict__ Wih, const u16* __restrict__ Whh,
        const float* __restrict__ bih, const float* __restrict__ bhh,
        const u16* __restrict__ Wres, const float* __restrict__ bres,
        float* __restrict__ hf, u16* __restrict__ hb, int t) {
    __shared__ float xg[16][SEG_ + 1];   // gathered segment
    __shared__ float xtf[16][D_];        // embedded segment, f32 (residual add)
    __shared__ u16   xtb[16 * D_];       // embedded segment, bf16 (WMMA A)
    __shared__ u16   htb[16 * D_];       // h carry tile, bf16 (WMMA A, via async DMA)
    __shared__ u16   h1b[16 * D_];       // GRU-cell output, bf16 (WMMA A)

    const int tid = threadIdx.x, lane = tid & 31, wave = tid >> 5;
    const int r0 = blockIdx.x * 16;

    // phase -1: kick off async DMA of the 8KB h tile (16 x 256 bf16) into LDS.
    // One copy shared by all 8 waves; overlaps with phases 0-1 below.
    for (int i = tid; i < 512; i += 256) {              // 512 x 16B chunks
        const int row = i >> 5, ch = i & 31;
        async_g2l_b128(hb + (size_t)(r0 + row) * D_ + ch * 8, &htb[row * D_ + ch * 8]);
    }

    // phase 0: gather x segment, subtract seq_last (coalesced over c)
    for (int i = tid; i < 16 * SEG_; i += 256) {
        const int rl = i & 15, s = i >> 4, r = r0 + rl;
        const int b = r / C_, c = r % C_;
        const float* xb = x + (size_t)b * L_ * C_ + c;
        xg[rl][s] = xb[(size_t)(t * SEG_ + s) * C_] - xb[(size_t)(L_ - 1) * C_];
    }
    __syncthreads();

    // phase 1: valueEmbedding (48->256) + SiLU, VALU (0.5 GFLOP/step, cheap)
    {
        const int m = tid & 15, d0 = tid >> 4;
        for (int d = d0; d < D_; d += 16) {
            float acc = bemb[d];
            const float* wr = Wemb + d * SEG_;
            #pragma unroll 8
            for (int s = 0; s < SEG_; ++s) acc += xg[m][s] * wr[s];
            const float v = acc * sigmoidf_(acc);       // SiLU
            xtf[m][d] = v;
            xtb[m * D_ + d] = f2bf(v);
        }
    }
    wait_async0();           // h tile resident in LDS (this wave's requests done)
    __syncthreads();         // ... and visible across all waves

    // phase 2: gates. Wave w owns column blocks {w, w+8}; per block 4 f32 accs:
    //   r,z share Gi+Gh accumulation; n needs i_n and h_n separately.
    for (int jj = 0; jj < 2; ++jj) {
        const int col0 = (wave + jj * 8) * 16;
        v8f ar = {}, az = {}, ai = {}, ah = {};
        for (int kb = 0; kb < D_; kb += 32) {
            const v16bf aX = load_a16(xtb, D_, kb, lane);
            const v16bf aH = load_a16(htb, D_, kb, lane);
            v16bf bb;
            bb = load_b16(Wih, D_, col0,        kb, lane); ar = WMMA_BF16(aX, bb, ar);
            bb = load_b16(Whh, D_, col0,        kb, lane); ar = WMMA_BF16(aH, bb, ar);
            bb = load_b16(Wih, D_, 256 + col0,  kb, lane); az = WMMA_BF16(aX, bb, az);
            bb = load_b16(Whh, D_, 256 + col0,  kb, lane); az = WMMA_BF16(aH, bb, az);
            bb = load_b16(Wih, D_, 512 + col0,  kb, lane); ai = WMMA_BF16(aX, bb, ai);
            bb = load_b16(Whh, D_, 512 + col0,  kb, lane); ah = WMMA_BF16(aH, bb, ah);
        }
        const int n = lane & 15, mh = (lane >> 4) * 8;
        #pragma unroll
        for (int e = 0; e < 8; ++e) {
            const int m = e + mh, col = col0 + n;
            const float gr = sigmoidf_(ar[e] + bih[col] + bhh[col]);
            const float gz = sigmoidf_(az[e] + bih[256 + col] + bhh[256 + col]);
            const float gn = tanhf(ai[e] + bih[512 + col] + gr * (ah[e] + bhh[512 + col]));
            const float hold = hf[(size_t)(r0 + m) * D_ + col];   // f32 carry
            h1b[m * D_ + col] = f2bf((1.0f - gz) * gn + gz * hold);
        }
    }
    __syncthreads();

    // phase 3: residual projection -> new carry (f32 + bf16 copies)
    for (int jj = 0; jj < 2; ++jj) {
        const int col0 = (wave + jj * 8) * 16;
        v8f acc = {};
        for (int kb = 0; kb < D_; kb += 32) {
            const v16bf aA = load_a16(h1b, D_, kb, lane);
            const v16bf bB = load_b16(Wres, D_, col0, kb, lane);
            acc = WMMA_BF16(aA, bB, acc);
        }
        const int n = lane & 15, mh = (lane >> 4) * 8;
        #pragma unroll
        for (int e = 0; e < 8; ++e) {
            const int m = e + mh, col = col0 + n;
            const float hn_ = xtf[m][col] + acc[e] + bres[col];
            hf[(size_t)(r0 + m) * D_ + col] = hn_;
            hb[(size_t)(r0 + m) * D_ + col] = f2bf(hn_);
        }
    }
}

// ---------------- kernel 2: generic WMMA GEMM + bias, out = A*W^T + bias -----
// grid = (N/16, M/128); block = 256 threads (8 waves, one 16x16 tile each).
// K must be <= 256 (K == 256 for all launches here); A tile staged via async DMA.
__global__ void __launch_bounds__(256) gemm_bias_kernel(
        const u16* __restrict__ A, const u16* __restrict__ W, const float* __restrict__ bias,
        float* __restrict__ out, int K, int M) {
    __shared__ u16 atile[16 * D_];
    const int lane = threadIdx.x & 31, wave = threadIdx.x >> 5;
    const int row0 = blockIdx.x * 16;
    const int col0 = (blockIdx.y * 8 + wave) * 16;
    __builtin_prefetch(W + (size_t)col0 * K, 0, 0);   // global_prefetch_b8 (L2-resident weights)

    const int cpr = K >> 3;                            // 16B chunks per A row
    for (int i = threadIdx.x; i < 16 * cpr; i += 256) {
        const int row = i / cpr, ch = i % cpr;
        async_g2l_b128(A + (size_t)(row0 + row) * K + ch * 8, &atile[row * K + ch * 8]);
    }
    wait_async0();
    __syncthreads();

    v8f acc = {};
    for (int kb = 0; kb < K; kb += 32) {
        const v16bf a = load_a16(atile, K, kb, lane);
        const v16bf b = load_b16(W, K, col0, kb, lane);
        acc = WMMA_BF16(a, b, acc);
    }
    const int n = lane & 15, mh = (lane >> 4) * 8;
    #pragma unroll
    for (int e = 0; e < 8; ++e)
        out[(size_t)(row0 + e + mh) * M + col0 + n] = acc[e] + bias[col0 + n];
}

// ---------------- kernel 3: decoder gates + prediction head, one wave/(b,c) --
__global__ void __launch_bounds__(32) dec_kernel(
        const float* __restrict__ x, const float* __restrict__ Ti, const float* __restrict__ Gh2,
        const float* __restrict__ hf, const u16* __restrict__ Wpred,
        const float* __restrict__ bpred, float* __restrict__ out) {
    __shared__ u16 hyb[16 * D_];                 // 15 hy rows + 1 zero pad row
    const int lane = threadIdx.x;
    const int r = blockIdx.x, b = r / C_, c = r % C_;
    const float* gh = Gh2 + (size_t)r * G3_;     // gbhh already folded in
    const float* hn = hf + (size_t)r * D_;
    const float* tbase = Ti + (size_t)(c * SY_) * G3_;  // gbih already folded in
    for (int i = lane; i < 16 * D_; i += 32) {
        const int sy = i >> 8, d = i & 255;
        float v = 0.0f;
        if (sy < SY_) {
            const float* ti = tbase + (size_t)sy * G3_;
            const float gr = sigmoidf_(ti[d] + gh[d]);
            const float gz = sigmoidf_(ti[256 + d] + gh[256 + d]);
            const float gn = tanhf(ti[512 + d] + gr * gh[512 + d]);
            v = (1.0f - gz) * gn + gz * hn[d];
        }
        hyb[i] = f2bf(v);
    }
    __syncthreads();
    const float last = x[((size_t)b * L_ + (L_ - 1)) * C_ + c];
    float* ob = out + (size_t)b * (720 * C_) + c;
    for (int nt = 0; nt < 3; ++nt) {             // 48 output cols = 3 tiles
        v8f acc = {};
        for (int kb = 0; kb < D_; kb += 32) {
            const v16bf a  = load_a16(hyb, D_, kb, lane);
            const v16bf bb = load_b16(Wpred, D_, nt * 16, kb, lane);
            acc = WMMA_BF16(a, bb, acc);
        }
        const int n = lane & 15, mh = (lane >> 4) * 8;
        #pragma unroll
        for (int e = 0; e < 8; ++e) {
            const int sy = e + mh;
            if (sy < SY_) {
                const int s = nt * 16 + n;
                ob[(size_t)(sy * SEG_ + s) * C_] = acc[e] + bpred[s] + last;
            }
        }
    }
}

// ---------------- workspace layout (bytes) -----------------------------------
#define OFF_WIH    0ull
#define OFF_WHH    393216ull
#define OFF_WRES   786432ull
#define OFF_GWIH   917504ull
#define OFF_GWHH   1310720ull
#define OFF_WPRED  1703936ull
#define OFF_TOK    1728512ull      // 4816*256 bf16
#define OFF_HF     4194304ull      // 20544*256 f32
#define OFF_HB     25231360ull     // 20544*256 bf16
#define OFF_TI     35749888ull     // 4816*768  f32
#define OFF_GH2    50544640ull     // 20544*768 f32
// total = 113,655,808 bytes

extern "C" void kernel_launch(void* const* d_in, const int* in_sizes, int n_in,
                              void* d_out, int out_size, void* d_ws, size_t ws_size,
                              hipStream_t stream) {
    (void)in_sizes; (void)n_in; (void)out_size; (void)ws_size;
    const float* x        = (const float*)d_in[0];
    const float* Wemb     = (const float*)d_in[1];
    const float* bemb     = (const float*)d_in[2];
    const float* Wih      = (const float*)d_in[3];
    const float* Whh      = (const float*)d_in[4];
    const float* bih      = (const float*)d_in[5];
    const float* bhh      = (const float*)d_in[6];
    const float* Wres     = (const float*)d_in[7];
    const float* bres     = (const float*)d_in[8];
    const float* gWih     = (const float*)d_in[9];
    const float* gWhh     = (const float*)d_in[10];
    const float* gbih     = (const float*)d_in[11];
    const float* gbhh     = (const float*)d_in[12];
    const float* pos_emb  = (const float*)d_in[13];
    const float* chan_emb = (const float*)d_in[14];
    const float* Wpred    = (const float*)d_in[15];
    const float* bpred    = (const float*)d_in[16];
    float* out = (float*)d_out;

    char* ws = (char*)d_ws;
    u16*   Wih_bf   = (u16*)  (ws + OFF_WIH);
    u16*   Whh_bf   = (u16*)  (ws + OFF_WHH);
    u16*   Wres_bf  = (u16*)  (ws + OFF_WRES);
    u16*   gWih_bf  = (u16*)  (ws + OFF_GWIH);
    u16*   gWhh_bf  = (u16*)  (ws + OFF_GWHH);
    u16*   Wpred_bf = (u16*)  (ws + OFF_WPRED);
    u16*   tok_bf   = (u16*)  (ws + OFF_TOK);
    float* hf       = (float*)(ws + OFF_HF);
    u16*   hb       = (u16*)  (ws + OFF_HB);
    float* Ti       = (float*)(ws + OFF_TI);
    float* Gh2      = (float*)(ws + OFF_GH2);

    // 0) weights -> bf16, build decoder tokens; zero GRU carry
    prep_kernel<<<1024, 256, 0, stream>>>(Wih, Whh, Wres, gWih, gWhh, Wpred,
                                          pos_emb, chan_emb,
                                          Wih_bf, Whh_bf, Wres_bf,
                                          gWih_bf, gWhh_bf, Wpred_bf, tok_bf);
    zero_kernel<<<1024, 256, 0, stream>>>((uint4*)(ws + OFF_HF),
                                          (int)((OFF_TI - OFF_HF) / 16));

    // 1) sequential encoder scan: 15 fused embed+GRUCell+residual steps
    for (int t = 0; t < SX_; ++t)
        step_kernel<<<BC_ / 16, 256, 0, stream>>>(x, Wemb, bemb, Wih_bf, Whh_bf,
                                                  bih, bhh, Wres_bf, bres, hf, hb, t);

    // 2) decoder input-path GEMM over 4815 unique tokens (padded to 4816)
    gemm_bias_kernel<<<dim3(4816 / 16, G3_ / 128), 256, 0, stream>>>(
        tok_bf, gWih_bf, gbih, Ti, D_, G3_);
    // 3) decoder hidden-path GEMM over the 20544 unique hn rows
    gemm_bias_kernel<<<dim3(BC_ / 16, G3_ / 128), 256, 0, stream>>>(
        hb, gWhh_bf, gbhh, Gh2, D_, G3_);

    // 4) decoder gates + 256->48 prediction head + transposed scatter
    dec_kernel<<<BC_, 32, 0, stream>>>(x, Ti, Gh2, hf, Wpred_bf, bpred, out);
}
